// create_As_84189948936752
// MI455X (gfx1250) — matrix operand order repaired and checked by reading, not compiled, not run on hardware
//
#include <hip/hip_runtime.h>

typedef __attribute__((ext_vector_type(16))) _Float16 v16h;
typedef __attribute__((ext_vector_type(8)))  _Float16 v8h;
typedef __attribute__((ext_vector_type(2)))  _Float16 v2h;
typedef __attribute__((ext_vector_type(8)))  float    v8f;
typedef __attribute__((ext_vector_type(4)))  float    v4f;

#define KCH   256     // channels
#define NB    64      // neighbours per tile
#define CPADH 264     // padded row length (halves) of transposed h tile in LDS

// ---------------------------------------------------------------------------
// 512 threads = 16 waves. Wave w owns output channels [16w, 16w+16).
//   hp = W @ h                        via v_wmma_f32_16x16x32_f16 (K-chain of 8)
//   A += sin((k+1)d)/d * hp * u u^T   via rotation recurrence + FMA epilogue
// Two-level pipelining:
//   - global loads for tile t+1 issued before tile t's WMMA chain (VGPR d-buf)
//   - B fragments streamed from LDS with a depth-2 rotating buffer so each
//     WMMA waits on a ds_load issued two WMMAs earlier, not the previous one
// ---------------------------------------------------------------------------
__global__ __launch_bounds__(512)
void create_As_main_kernel(const float* __restrict__ h,
                           const float* __restrict__ rel,
                           const float* __restrict__ W,
                           float* __restrict__ ws,
                           int N, int nTiles) {
  __shared__ __align__(16) _Float16 ht[NB * CPADH];   // h tile, transposed [i][c], f16
  __shared__ float sD[NB], sInv[NB], sUx[NB], sUy[NB], sUz[NB];

  const int t    = threadIdx.x;
  const int lane = t & 31;
  const int wv   = t >> 5;       // wave id 0..15
  const int l16  = lane & 15;
  const int hf   = lane >> 4;    // half-wave 0/1
  const int kT   = wv * 16;      // this wave's channel-tile base

  // staging coordinates: 2048 units = (c-pair 0..127) x (i-quad 0..15)
  int ucp[4], ui4[4];
#pragma unroll
  for (int u = 0; u < 4; ++u) {
    int idx = t + 512 * u;
    ucp[u] = idx >> 4;           // channel pair -> c = 2*ucp, 2*ucp+1
    ui4[u] = idx & 15;           // neighbour quad -> i = 4*ui4 .. +3
  }

  // ---- W fragments (A-matrix), converted once, resident in VGPRs.
  // ISA 16-bit A 16x32 layout: lane row M=l16; halves 0..7 <- K = 8*hf+0..7,
  // halves 8..15 <- K = 16+8*hf+0..7 (per K-step base kk*32).
  const float* Wrow = W + (size_t)(kT + l16) * KCH;
  v16h wfrag[8];
#pragma unroll
  for (int kk = 0; kk < 8; ++kk) {
    const float* p = Wrow + kk * 32 + 8 * hf;
    v4f A0 = *(const v4f*)(p);
    v4f A1 = *(const v4f*)(p + 4);
    v4f B0 = *(const v4f*)(p + 16);
    v4f B1 = *(const v4f*)(p + 20);
    v16h a = {};
    a[0]  = (_Float16)A0[0]; a[1]  = (_Float16)A0[1]; a[2]  = (_Float16)A0[2]; a[3]  = (_Float16)A0[3];
    a[4]  = (_Float16)A1[0]; a[5]  = (_Float16)A1[1]; a[6]  = (_Float16)A1[2]; a[7]  = (_Float16)A1[3];
    a[8]  = (_Float16)B0[0]; a[9]  = (_Float16)B0[1]; a[10] = (_Float16)B0[2]; a[11] = (_Float16)B0[3];
    a[12] = (_Float16)B1[0]; a[13] = (_Float16)B1[1]; a[14] = (_Float16)B1[2]; a[15] = (_Float16)B1[3];
    wfrag[kk] = a;
  }

  float acc[8][6];
#pragma unroll
  for (int r = 0; r < 8; ++r)
#pragma unroll
    for (int q = 0; q < 6; ++q) acc[r][q] = 0.0f;

  v4f preA[4], preB[4];          // in-flight h tile (two channel rows per unit)
  float px = 0.f, py = 0.f, pz = 0.f;

  // ---- prologue: fetch + stage first tile
  int tile = blockIdx.x;
  {
    const int i0 = tile * NB;
#pragma unroll
    for (int u = 0; u < 4; ++u) {
      const float* pa = h + (size_t)(2 * ucp[u]) * N + i0 + 4 * ui4[u];
      preA[u] = *(const v4f*)pa;
      preB[u] = *(const v4f*)(pa + N);
    }
    if (t < NB) { px = rel[i0 + t]; py = rel[N + i0 + t]; pz = rel[2 * N + i0 + t]; }
  }
#pragma unroll
  for (int u = 0; u < 4; ++u)
#pragma unroll
    for (int q = 0; q < 4; ++q) {
      v2h pk = { (_Float16)preA[u][q], (_Float16)preB[u][q] };
      *(v2h*)(ht + (4 * ui4[u] + q) * CPADH + 2 * ucp[u]) = pk;
    }
  if (t < NB) {
    float dd = sqrtf(px * px + py * py + pz * pz);
    float iv = 1.0f / dd;
    sD[t] = dd; sInv[t] = iv; sUx[t] = px * iv; sUy[t] = py * iv; sUz[t] = pz * iv;
  }
  __syncthreads();

  // base pointer for this lane's B fragments
  const _Float16* bbase = ht + (size_t)l16 * CPADH + 8 * hf;

  for (; tile < nTiles; tile += gridDim.x) {
    // ---- issue next tile's global loads; they complete under the WMMA chain
    const int nxt = tile + (int)gridDim.x;
    const int i0n = (nxt < nTiles ? nxt : tile) * NB;
#pragma unroll
    for (int u = 0; u < 4; ++u) {
      const float* pa = h + (size_t)(2 * ucp[u]) * N + i0n + 4 * ui4[u];
      preA[u] = *(const v4f*)pa;
      preB[u] = *(const v4f*)(pa + N);
    }
    if (t < NB) { px = rel[i0n + t]; py = rel[N + i0n + t]; pz = rel[2 * N + i0n + t]; }

    // ---- GEMM over a flat stream of 32 fragments: j = (kk<<2) | n.
    // ISA 16-bit B 32x16 layout: lane col N=l16; halves 0..7 <- K = 8*hf+0..7, halves 8..15 <- K = 16+8*hf+0..7.
    // Depth-2 rotating buffer: fragment j+2 loads while fragment j is consumed.
    v8f C[4] = {};
    v16h bbuf[2];
#pragma unroll
    for (int j = 0; j < 2; ++j) {
      const _Float16* bp = bbase + (j & 3) * 16 * CPADH + (j >> 2) * 32;
      v8h blo = *(const v8h*)(bp);
      v8h bhi = *(const v8h*)(bp + 16);
      bbuf[j] = __builtin_shufflevector(blo, bhi,
                                        0, 1, 2, 3, 4, 5, 6, 7,
                                        8, 9, 10, 11, 12, 13, 14, 15);
    }
#pragma unroll
    for (int j = 0; j < 32; ++j) {
      v16h bnext = {};
      if (j + 2 < 32) {
        const int jn = j + 2;
        const _Float16* bp = bbase + (jn & 3) * 16 * CPADH + (jn >> 2) * 32;
        v8h blo = *(const v8h*)(bp);
        v8h bhi = *(const v8h*)(bp + 16);
        bnext = __builtin_shufflevector(blo, bhi,
                                        0, 1, 2, 3, 4, 5, 6, 7,
                                        8, 9, 10, 11, 12, 13, 14, 15);
      }
      const int n = j & 3;
      C[n] = __builtin_amdgcn_wmma_f32_16x16x32_f16(false, wfrag[j >> 2], false, bbuf[j & 1],
                                                    (short)0, C[n], false, false);
      if (j + 2 < 32) bbuf[j & 1] = bnext;
    }

    // ---- epilogue: C/D layout M = r + 8*hf, N = l16 -> per lane the
    // neighbour i is fixed across the 8 regs while k steps by 1; advance
    // sin((k+1)d) with a (cos d, sin d) rotation instead of 8 transcendentals.
#pragma unroll
    for (int n = 0; n < 4; ++n) {
      int il = n * 16 + l16;
      float Dd = sD[il], iv = sInv[il];
      float ux = sUx[il], uy = sUy[il], uz = sUz[il];
      float uu0 = ux * ux, uu1 = ux * uy, uu2 = ux * uz;
      float uu3 = uy * uy, uu4 = uy * uz, uu5 = uz * uz;
      float sd, cd; sincosf(Dd, &sd, &cd);
      float ph = (float)(kT + 8 * hf + 1) * Dd;
      float s, cv; sincosf(ph, &s, &cv);
#pragma unroll
      for (int r = 0; r < 8; ++r) {
        float tt = C[n][r] * s * iv;                // hp * sin((k+1)d) / d
        acc[r][0] = fmaf(tt, uu0, acc[r][0]);
        acc[r][1] = fmaf(tt, uu1, acc[r][1]);
        acc[r][2] = fmaf(tt, uu2, acc[r][2]);
        acc[r][3] = fmaf(tt, uu3, acc[r][3]);
        acc[r][4] = fmaf(tt, uu4, acc[r][4]);
        acc[r][5] = fmaf(tt, uu5, acc[r][5]);
        float s2 = fmaf(s, cd, cv * sd);            // rotate phase by d
        cv = fmaf(cv, cd, -(s * sd));
        s = s2;
      }
    }

    __syncthreads();   // all reads of current tile done
    // ---- convert + store the prefetched tile (transposed, packed b32)
#pragma unroll
    for (int u = 0; u < 4; ++u)
#pragma unroll
      for (int q = 0; q < 4; ++q) {
        v2h pk = { (_Float16)preA[u][q], (_Float16)preB[u][q] };
        *(v2h*)(ht + (4 * ui4[u] + q) * CPADH + 2 * ucp[u]) = pk;
      }
    if (t < NB) {
      float dd = sqrtf(px * px + py * py + pz * pz);
      float iv = 1.0f / dd;
      sD[t] = dd; sInv[t] = iv; sUx[t] = px * iv; sUy[t] = py * iv; sUz[t] = pz * iv;
    }
    __syncthreads();   // staged tile visible
  }

  // ---- reduce the 16 lanes sharing a k-set, fold into workspace
#pragma unroll
  for (int r = 0; r < 8; ++r) {
#pragma unroll
    for (int q = 0; q < 6; ++q) {
      float v = acc[r][q];
      v += __shfl_xor(v, 1);
      v += __shfl_xor(v, 2);
      v += __shfl_xor(v, 4);
      v += __shfl_xor(v, 8);
      if (l16 == 0) atomicAdd(&ws[(kT + 8 * hf + r) * 6 + q], v);
    }
  }
}

// ---------------------------------------------------------------------------
// Expand the 6 unique symmetric components into the full 256x3x3 output.
// ---------------------------------------------------------------------------
__global__ void create_As_finalize_kernel(const float* __restrict__ ws,
                                          float* __restrict__ out) {
  int k = threadIdx.x;   // 256 threads
  float xx = ws[k * 6 + 0], xy = ws[k * 6 + 1], xz = ws[k * 6 + 2];
  float yy = ws[k * 6 + 3], yz = ws[k * 6 + 4], zz = ws[k * 6 + 5];
  float* o = out + k * 9;
  o[0] = xx; o[1] = xy; o[2] = xz;
  o[3] = xy; o[4] = yy; o[5] = yz;
  o[6] = xz; o[7] = yz; o[8] = zz;
}

extern "C" void kernel_launch(void* const* d_in, const int* in_sizes, int n_in,
                              void* d_out, int out_size, void* d_ws, size_t ws_size,
                              hipStream_t stream) {
  const float* h   = (const float*)d_in[0];   // [256, N]
  const float* rel = (const float*)d_in[1];   // [3, N]
  const float* W   = (const float*)d_in[2];   // [256, 256]
  float* ws  = (float*)d_ws;                  // 256*6 f32 partials
  float* out = (float*)d_out;                 // 256*9 f32

  const int N = in_sizes[0] / KCH;            // 262144
  const int nTiles = N / NB;                  // 4096
  const int blocks = nTiles < 1024 ? nTiles : 1024;

  hipMemsetAsync(ws, 0, KCH * 6 * sizeof(float), stream);
  create_As_main_kernel<<<blocks, 512, 0, stream>>>(h, rel, W, ws, N, nTiles);
  create_As_finalize_kernel<<<1, KCH, 0, stream>>>(ws, out);
}
